// TtWhisperAttention_5497558139580
// MI455X (gfx1250) — compile-verified, matrix operand order checked by reading, not run on hardware
//
#include <hip/hip_runtime.h>

#define SEQ 1500
#define EMB 1024
#define NH  16
#define HD  64
#define MROWS 6000   // B*SEQ = 4*1500

typedef __attribute__((ext_vector_type(16))) _Float16 v16h;
typedef __attribute__((ext_vector_type(8)))  _Float16 v8h;
typedef __attribute__((ext_vector_type(8)))  float    v8f;

__device__ inline v8h zero8h() { v8h z = {}; return z; }
__device__ inline v8f zero8f() { v8f z = {}; return z; }

__device__ inline v16h make16(v8h lo, v8h hi) {
  v16h r;
#pragma unroll
  for (int i = 0; i < 8; ++i) { r[i] = lo[i]; r[i + 8] = hi[i]; }
  return r;
}

// Load a 16x32 f16 A/B WMMA fragment whose row (m or n) = lane%16 is already
// folded into rowp. Per ISA layout: k(h,lane) = (h/8)*16 + (lane/16)*8 + h%8,
// i.e. two contiguous 8-half (16B) segments at +((lane>>4)*8) and +16 more.
__device__ inline v16h ld_frag(const _Float16* rowp, int lane) {
  const _Float16* p = rowp + ((lane >> 4) << 3);
  v8h lo = *(const v8h*)(p);
  v8h hi = *(const v8h*)(p + 16);
  return make16(lo, hi);
}

__global__ void cvt_f32_to_f16(const float* __restrict__ s,
                               _Float16* __restrict__ d, int n) {
  int i = blockIdx.x * blockDim.x + threadIdx.x;
  int stride = gridDim.x * blockDim.x;
  for (; i < n; i += stride) d[i] = (_Float16)s[i];
}

// C[M,N] = (X[M,K] * W[N,K]^T + bias) * scale ; K = N = 1024.
// mode 0: write f16 into [b, h, t, d] head layout (QKV projections)
// mode 1: write f32 row-major [M, N] (output projection -> d_out)
// Ping-pong double-buffered LDS: one barrier per K-step; next tile's global
// loads are issued before the 8 WMMAs so load latency hides under compute.
__global__ __launch_bounds__(256) void gemm_xwT(
    const _Float16* __restrict__ X, const _Float16* __restrict__ W,
    const float* __restrict__ bias, float scale, int M, int mode,
    _Float16* __restrict__ outH, float* __restrict__ outF) {
  __shared__ _Float16 sA[2][128 * 40];  // 128 rows x 32 K, pad stride 40
  __shared__ _Float16 sB[2][128 * 40];

  const int tid  = threadIdx.x;
  const int lane = tid & 31;
  const int wave = tid >> 5;
  const int wm   = wave >> 1;  // 0..3 -> 32 rows each
  const int wn   = wave & 1;   // 0..1 -> 64 cols each
  const int mblk = blockIdx.x * 128;
  const int nblk = blockIdx.y * 128;

  v8f acc[2][4];
#pragma unroll
  for (int i = 0; i < 2; ++i)
#pragma unroll
    for (int j = 0; j < 4; ++j) acc[i][j] = zero8f();

  const int lrow = tid >> 1;        // 0..127
  const int lcol = (tid & 1) * 16;  // 0 or 16
  const int lidx = lrow * 40 + lcol;

  // Clamp tail rows instead of EXEC-masking: keeps copies uniform; clamped
  // rows only feed C rows >= M which are never stored.
  int gm = mblk + lrow;
  if (gm >= M) gm = M - 1;
  const _Float16* gA = X + (size_t)gm * EMB + lcol;
  const _Float16* gB = W + (size_t)(nblk + lrow) * EMB + lcol;  // N always ok

  // prologue: stage kb = 0 into buffer 0
  {
    v8h a0 = *(const v8h*)(gA);
    v8h a1 = *(const v8h*)(gA + 8);
    v8h b0 = *(const v8h*)(gB);
    v8h b1 = *(const v8h*)(gB + 8);
    *(v8h*)&sA[0][lidx]     = a0;
    *(v8h*)&sA[0][lidx + 8] = a1;
    *(v8h*)&sB[0][lidx]     = b0;
    *(v8h*)&sB[0][lidx + 8] = b1;
  }
  __syncthreads();

  int cur = 0;
  for (int kb = 0; kb < EMB; kb += 32, cur ^= 1) {
    const int  nxt  = cur ^ 1;
    const bool more = (kb + 32) < EMB;

    // issue next tile's global loads early (in-flight during WMMAs)
    v8h a0 = zero8h(), a1 = zero8h(), b0 = zero8h(), b1 = zero8h();
    if (more) {
      a0 = *(const v8h*)(gA + kb + 32);
      a1 = *(const v8h*)(gA + kb + 40);
      b0 = *(const v8h*)(gB + kb + 32);
      b1 = *(const v8h*)(gB + kb + 40);
    }

    // compute on current buffer
    v16h aF[2];
#pragma unroll
    for (int mi = 0; mi < 2; ++mi)
      aF[mi] = ld_frag(&sA[cur][(wm * 32 + mi * 16 + (lane & 15)) * 40], lane);
#pragma unroll
    for (int ni = 0; ni < 4; ++ni) {
      v16h bF = ld_frag(&sB[cur][(wn * 64 + ni * 16 + (lane & 15)) * 40], lane);
#pragma unroll
      for (int mi = 0; mi < 2; ++mi)
        acc[mi][ni] = __builtin_amdgcn_wmma_f32_16x16x32_f16(
            false, aF[mi], false, bF, (short)0, acc[mi][ni], false, false);
    }

    // stage next tile into the other buffer
    if (more) {
      *(v8h*)&sA[nxt][lidx]     = a0;
      *(v8h*)&sA[nxt][lidx + 8] = a1;
      *(v8h*)&sB[nxt][lidx]     = b0;
      *(v8h*)&sB[nxt][lidx + 8] = b1;
    }
    __syncthreads();
  }

  // epilogue: C layout -> element (m = r + 8*(lane/16), n = lane%16) per frag
#pragma unroll
  for (int mi = 0; mi < 2; ++mi) {
#pragma unroll
    for (int ni = 0; ni < 4; ++ni) {
#pragma unroll
      for (int r = 0; r < 8; ++r) {
        int m = mblk + wm * 32 + mi * 16 + r + 8 * (lane >> 4);
        int n = nblk + wn * 64 + ni * 16 + (lane & 15);
        if (m < M) {
          float v = acc[mi][ni][r];
          if (bias) v += bias[n];
          v *= scale;
          if (mode == 0) {
            int b = m / SEQ, t = m % SEQ;
            int h = n >> 6, d = n & 63;
            outH[(((size_t)(b * NH + h) * SEQ) + t) * HD + d] = (_Float16)v;
          } else {
            outF[(size_t)m * EMB + n] = v;
          }
        }
      }
    }
  }
}

// Flash attention over Q,K,V in [b*h, t, d] f16. Output f16 in [b, t, e].
__global__ __launch_bounds__(256) void attn_kernel(
    const _Float16* __restrict__ Q, const _Float16* __restrict__ K,
    const _Float16* __restrict__ V, _Float16* __restrict__ O) {
  __shared__ _Float16 kt[64][72];     // [kv][d]
  __shared__ _Float16 vt[64][72];     // [d][kv]  (transposed)
  __shared__ _Float16 pb[8][16][72];  // per-wave P scratch [row][kv]

  const int tid  = threadIdx.x;
  const int lane = tid & 31;
  const int w    = tid >> 5;
  const int bh   = blockIdx.y;
  const int q0   = blockIdx.x * 128 + w * 16;

  const _Float16* Qb = Q + (size_t)bh * SEQ * HD;
  const _Float16* Kb = K + (size_t)bh * SEQ * HD;
  const _Float16* Vb = V + (size_t)bh * SEQ * HD;

  // Q fragments for this wave's 16 rows (clamped; invalid rows never stored)
  int qrow = q0 + (lane & 15);
  int qr = qrow < SEQ ? qrow : (SEQ - 1);
  v16h qf[2];
#pragma unroll
  for (int s = 0; s < 2; ++s)
    qf[s] = ld_frag(Qb + (size_t)qr * HD + s * 32, lane);

  v8f oacc[4];
#pragma unroll
  for (int c = 0; c < 4; ++c) oacc[c] = zero8f();
  float mrow[8], lrow[8];
#pragma unroll
  for (int r = 0; r < 8; ++r) { mrow[r] = -1e30f; lrow[r] = 0.f; }

  const int lkv = tid >> 2;        // 0..63
  const int ld0 = (tid & 3) * 16;  // 0,16,32,48

  for (int kv0 = 0; kv0 < SEQ; kv0 += 64) {
    __syncthreads();
    int kvr = kv0 + lkv;
    v8h k0 = zero8h(), k1 = zero8h(), v0 = zero8h(), v1 = zero8h();
    if (kvr < SEQ) {
      k0 = *(const v8h*)(Kb + (size_t)kvr * HD + ld0);
      k1 = *(const v8h*)(Kb + (size_t)kvr * HD + ld0 + 8);
      v0 = *(const v8h*)(Vb + (size_t)kvr * HD + ld0);
      v1 = *(const v8h*)(Vb + (size_t)kvr * HD + ld0 + 8);
    }
    *(v8h*)&kt[lkv][ld0]     = k0;
    *(v8h*)&kt[lkv][ld0 + 8] = k1;
#pragma unroll
    for (int j = 0; j < 8; ++j) {
      vt[ld0 + j][lkv]     = v0[j];
      vt[ld0 + 8 + j][lkv] = v1[j];
    }
    __syncthreads();

    // S = Q K^T : 4 kv-chunks x 2 k-steps
    v8f sa[4];
#pragma unroll
    for (int c = 0; c < 4; ++c) sa[c] = zero8f();
#pragma unroll
    for (int c = 0; c < 4; ++c) {
#pragma unroll
      for (int s = 0; s < 2; ++s) {
        v16h bF = ld_frag(&kt[c * 16 + (lane & 15)][s * 32], lane);
        sa[c] = __builtin_amdgcn_wmma_f32_16x16x32_f16(
            false, qf[s], false, bF, (short)0, sa[c], false, false);
      }
    }

    // mask OOB keys + new row max
    float nmax[8];
#pragma unroll
    for (int r = 0; r < 8; ++r) nmax[r] = mrow[r];
#pragma unroll
    for (int c = 0; c < 4; ++c) {
      int ng = kv0 + c * 16 + (lane & 15);
#pragma unroll
      for (int r = 0; r < 8; ++r) {
        float sc = (ng < SEQ) ? sa[c][r] : -1e30f;
        sa[c][r] = sc;
        nmax[r] = fmaxf(nmax[r], sc);
      }
    }
#pragma unroll
    for (int off = 1; off < 16; off <<= 1)
#pragma unroll
      for (int r = 0; r < 8; ++r)
        nmax[r] = fmaxf(nmax[r], __shfl_xor(nmax[r], off, 32));

    float fac[8];
#pragma unroll
    for (int r = 0; r < 8; ++r) {
      fac[r] = __expf(mrow[r] - nmax[r]);
      mrow[r] = nmax[r];
      lrow[r] *= fac[r];
    }
#pragma unroll
    for (int c = 0; c < 4; ++c)
#pragma unroll
      for (int r = 0; r < 8; ++r) oacc[c][r] *= fac[r];

    // P = exp(S - m), accumulate row sums, spill P to LDS in [row][kv]
    float psum[8];
#pragma unroll
    for (int r = 0; r < 8; ++r) psum[r] = 0.f;
#pragma unroll
    for (int c = 0; c < 4; ++c) {
#pragma unroll
      for (int r = 0; r < 8; ++r) {
        float p = __expf(sa[c][r] - mrow[r]);
        psum[r] += p;
        pb[w][r + 8 * (lane >> 4)][c * 16 + (lane & 15)] = (_Float16)p;
      }
    }
#pragma unroll
    for (int off = 1; off < 16; off <<= 1)
#pragma unroll
      for (int r = 0; r < 8; ++r) psum[r] += __shfl_xor(psum[r], off, 32);
#pragma unroll
    for (int r = 0; r < 8; ++r) lrow[r] += psum[r];

    // O += P V : same-wave LDS store->load is in-order
    v16h pF[2];
#pragma unroll
    for (int s = 0; s < 2; ++s)
      pF[s] = ld_frag(&pb[w][lane & 15][s * 32], lane);
#pragma unroll
    for (int c = 0; c < 4; ++c) {
#pragma unroll
      for (int s = 0; s < 2; ++s) {
        v16h vF = ld_frag(&vt[c * 16 + (lane & 15)][s * 32], lane);
        oacc[c] = __builtin_amdgcn_wmma_f32_16x16x32_f16(
            false, pF[s], false, vF, (short)0, oacc[c], false, false);
      }
    }
  }

  int b = bh >> 4, h = bh & 15;
#pragma unroll
  for (int c = 0; c < 4; ++c) {
#pragma unroll
    for (int r = 0; r < 8; ++r) {
      int m = q0 + r + 8 * (lane >> 4);
      if (m < SEQ) {
        float o = oacc[c][r] / lrow[r];
        O[((size_t)(b * SEQ + m)) * EMB + h * HD + c * 16 + (lane & 15)] =
            (_Float16)o;
      }
    }
  }
}

extern "C" void kernel_launch(void* const* d_in, const int* in_sizes, int n_in,
                              void* d_out, int out_size, void* d_ws,
                              size_t ws_size, hipStream_t stream) {
  const float* x  = (const float*)d_in[0];
  const float* Wq = (const float*)d_in[1];
  const float* bq = (const float*)d_in[2];
  const float* Wk = (const float*)d_in[3];
  const float* Wv = (const float*)d_in[4];
  const float* bv = (const float*)d_in[5];
  const float* Wo = (const float*)d_in[6];
  const float* bo = (const float*)d_in[7];
  float* out = (float*)d_out;

  _Float16* base = (_Float16*)d_ws;
  size_t off = 0;
  auto carve = [&](size_t n) {
    _Float16* p = base + off;
    off += (n + 127) & ~(size_t)127;
    return p;
  };
  _Float16* Xh  = carve((size_t)MROWS * EMB);
  _Float16* Wqh = carve((size_t)EMB * EMB);
  _Float16* Wkh = carve((size_t)EMB * EMB);
  _Float16* Wvh = carve((size_t)EMB * EMB);
  _Float16* Woh = carve((size_t)EMB * EMB);
  _Float16* Qh  = carve((size_t)MROWS * EMB);
  _Float16* Kh  = carve((size_t)MROWS * EMB);
  _Float16* Vh  = carve((size_t)MROWS * EMB);
  _Float16* Ah  = carve((size_t)MROWS * EMB);

  cvt_f32_to_f16<<<512, 256, 0, stream>>>(x,  Xh,  MROWS * EMB);
  cvt_f32_to_f16<<<512, 256, 0, stream>>>(Wq, Wqh, EMB * EMB);
  cvt_f32_to_f16<<<512, 256, 0, stream>>>(Wk, Wkh, EMB * EMB);
  cvt_f32_to_f16<<<512, 256, 0, stream>>>(Wv, Wvh, EMB * EMB);
  cvt_f32_to_f16<<<512, 256, 0, stream>>>(Wo, Woh, EMB * EMB);

  dim3 g((MROWS + 127) / 128, EMB / 128);  // 47 x 8
  // q = (X Wq^T + bq) * d^-0.5  (scale applied after bias, as in reference)
  gemm_xwT<<<g, 256, 0, stream>>>(Xh, Wqh, bq, 0.125f, MROWS, 0, Qh, nullptr);
  gemm_xwT<<<g, 256, 0, stream>>>(Xh, Wkh, nullptr, 1.0f, MROWS, 0, Kh, nullptr);
  gemm_xwT<<<g, 256, 0, stream>>>(Xh, Wvh, bv, 1.0f, MROWS, 0, Vh, nullptr);

  attn_kernel<<<dim3((SEQ + 127) / 128, 4 * NH), 256, 0, stream>>>(Qh, Kh, Vh,
                                                                   Ah);

  gemm_xwT<<<g, 256, 0, stream>>>(Ah, Woh, bo, 1.0f, MROWS, 1, nullptr, out);
}
